// FMDMoveScorer_56813827392047
// MI455X (gfx1250) — matrix operand order repaired and checked by reading, not compiled, not run on hardware
//
#include <hip/hip_runtime.h>
#include <hip/hip_bf16.h>
#include <math.h>
#include <stdint.h>

// ---------------- problem constants ----------------
constexpr int B_ = 32, N_ = 2048, F_ = 32, E_ = 16384, M_ = 2048, D_ = 128, L_ = 4;

typedef __bf16 bf16_t;
typedef __attribute__((ext_vector_type(16))) __bf16 v16bf;
typedef __attribute__((ext_vector_type(8)))  __bf16 v8bf;
typedef __attribute__((ext_vector_type(8)))  float  v8f;

union ABf { v16bf v; v8bf h[2]; };

// Force global (address_space(1)) accesses so the compiler emits
// global_load_*/global_store_* instead of flat_* (flat burns DScnt too).
template <typename T>
__device__ __forceinline__ T gload(const void* p) {
    return *(const __attribute__((address_space(1))) T*)(uintptr_t)p;
}
template <typename T>
__device__ __forceinline__ void gstore(void* p, T v) {
    *(__attribute__((address_space(1))) T*)(uintptr_t)p = v;
}

__device__ __forceinline__ float silu_f(float x) {
    return x / (1.f + __expf(-x));
}

// ---------------- small helper kernels ----------------

// t_emb = silu(time_encoding(t) @ Wt1 + bt1) @ Wt2 + bt2   -> (B,128) f32
__global__ void time_embed_kernel(const float* __restrict__ t,
                                  const float* __restrict__ Wt1, const float* __restrict__ bt1,
                                  const float* __restrict__ Wt2, const float* __restrict__ bt2,
                                  float* __restrict__ t_emb) {
    __shared__ float h1[B_][D_];
    const int j = threadIdx.x;                 // 0..127
    const float c = -9.210340371976184f / 8.f; // -ln(10000)/8
    for (int b = 0; b < B_; ++b) {
        float tv = t[b];
        float s = bt1[j];
        #pragma unroll
        for (int k = 0; k < 16; ++k) {
            float fr = __expf((float)(k & 7) * c);
            float te = (k < 8) ? __sinf(tv * fr) : __cosf(tv * fr);
            s += te * Wt1[k * D_ + j];
        }
        h1[b][j] = silu_f(s);
    }
    __syncthreads();
    for (int b = 0; b < B_; ++b) {
        float s = bt2[j];
        for (int k = 0; k < D_; ++k) s += h1[b][k] * Wt2[k * D_ + j];
        t_emb[b * D_ + j] = s;
    }
}

// WT[n*K + k] = (bf16) W[k*N + n]   (W is K x N row-major)
__global__ void transpose_bf16_kernel(const float* __restrict__ W, bf16_t* __restrict__ WT,
                                      int K, int Nn) {
    int i = blockIdx.x * 256 + threadIdx.x;
    if (i < K * Nn) {
        int k = i / Nn, n = i % Nn;
        gstore<bf16_t>(WT + (size_t)n * K + k, (bf16_t)gload<float>(W + i));
    }
}

__global__ void cvt_bf16_kernel(const float* __restrict__ x, bf16_t* __restrict__ y, int n) {
    int i = blockIdx.x * 256 + threadIdx.x;
    if (i < n) gstore<bf16_t>(y + i, (bf16_t)gload<float>(x + i));
}

__global__ void zero_kernel(float* __restrict__ x, int n) {
    int i = blockIdx.x * 256 + threadIdx.x;
    if (i < n) gstore<float>(x + i, 0.f);
}

// scores[r] = mask[r] ? dot(s2[r,:], Ws3) + bs3 : -inf   (one wave per row)
__global__ void score_kernel(const bf16_t* __restrict__ s2, const float* __restrict__ Ws3,
                             const float* __restrict__ bs3, const unsigned char* __restrict__ mask,
                             float* __restrict__ out, int rows) {
    int gw = (blockIdx.x * 256 + (int)threadIdx.x) >> 5;
    int lane = threadIdx.x & 31;
    if (gw >= rows) return;
    const bf16_t* rp = s2 + (size_t)gw * D_ + lane * 4;
    float s = 0.f;
    #pragma unroll
    for (int i = 0; i < 4; ++i) s += (float)gload<bf16_t>(rp + i) * Ws3[lane * 4 + i];
    #pragma unroll
    for (int off = 16; off > 0; off >>= 1) s += __shfl_xor(s, off, 32);
    if (lane == 0) gstore<float>(out + gw, mask[gw] ? (s + bs3[0]) : -INFINITY);
}

// ---------------- the unified WMMA GEMM kernel ----------------
// Out(rows x 128) = f(A(rows x K) @ W(K x 128)). W passed transposed:
// 128 x K row-major bf16, so every B fragment is one contiguous 32B lane load.
// 256 threads = 8 waves; wave computes 16 rows x 128 cols = 8 wmma tiles.
// K is a compile-time template param -> fully unrolled k-loop -> the
// scheduler hoists the next step's loads above the current WMMAs.

enum { GM_EMBED1 = 0, GM_EMBED2, GM_MSG, GM_UPD, GM_SCORE1, GM_SCORE2 };

struct GemmArgs {
    const bf16_t* A0;    // primary activation base (bf16)
    const bf16_t* A1;    // secondary base (agg) for UPD
    const bf16_t* W;     // transposed weight, 128 x K row-major
    const float*  bias;  // (128) f32
    const float*  extra; // t_emb (B x 128) for EMBED2
    const int*    src;   // edge src (E) for MSG
    const int*    dst;   // edge dst (E) for MSG
    const int*    mn;    // move_nodes (rows x 4) for SCORE1
    float*        outF;  // f32 output (h / agg atomic target)
    bf16_t*       outB;  // bf16 output
};

template <int MODE, int KSTEPS>
__launch_bounds__(256)
__global__ void gemm_kernel(GemmArgs p) {
    constexpr int K = KSTEPS * 32;
    const int lane = threadIdx.x & 31;
    const int wave = threadIdx.x >> 5;
    const int lmod = lane & 15;   // row-in-tile for A, col-in-tile for B/D
    const int lhi  = lane >> 4;   // K-half select for A/B, row-half for D
    const int rowTile = blockIdx.x * 128 + wave * 16;
    const int rA = rowTile + lmod;

    // per-lane base pointers of the 128-wide A segments for row rA
    const bf16_t* seg[4];
    if constexpr (MODE == GM_EMBED1) {
        seg[0] = p.A0 + (size_t)rA * 32;                    // K = 32 (node features)
    } else if constexpr (MODE == GM_EMBED2 || MODE == GM_SCORE2) {
        seg[0] = p.A0 + (size_t)rA * 128;
    } else if constexpr (MODE == GM_MSG) {
        const int b = rA / E_, e = rA % E_;
        seg[0] = p.A0 + ((size_t)b * N_ + gload<int>(p.src + e)) * 128;  // h[b, src[e]]
        seg[1] = p.A0 + ((size_t)b * N_ + gload<int>(p.dst + e)) * 128;  // h[b, dst[e]]
    } else if constexpr (MODE == GM_UPD) {
        seg[0] = p.A0 + (size_t)rA * 128;                   // h row
        seg[1] = p.A1 + (size_t)rA * 128;                   // agg row
    } else {                                                // SCORE1: 4-node gather
        const int b = rA / M_;
        #pragma unroll
        for (int q = 0; q < 4; ++q) {
            int idx = gload<int>(p.mn + (size_t)rA * 4 + q);
            idx = idx < 0 ? 0 : (idx > N_ - 1 ? N_ - 1 : idx);
            seg[q] = p.A0 + ((size_t)b * N_ + idx) * 128;
        }
    }

    v8f acc[8] = {};
    #pragma unroll
    for (int ks = 0; ks < KSTEPS; ++ks) {
        // A fragment: lane lmod holds row rA; 16 contiguous K=[kb+16*lhi..kb+16*lhi+15]
        const bf16_t* ap = seg[ks >> 2] + ((ks & 3) << 5) + (lhi << 4);
        ABf a;
        a.h[0] = gload<v8bf>(ap);
        a.h[1] = gload<v8bf>(ap + 8);
        // B fragments: lane holds column n = nt*16+lmod, 16 contiguous K (32B)
        const bf16_t* wb = p.W + (size_t)lmod * K + (ks << 5) + (lhi << 4);
        v16bf bfr[8];
        #pragma unroll
        for (int nt = 0; nt < 8; ++nt)
            bfr[nt] = gload<v16bf>(wb + (size_t)(nt * 16) * K);
        #pragma unroll
        for (int nt = 0; nt < 8; ++nt)
            acc[nt] = __builtin_amdgcn_wmma_f32_16x16x32_bf16(
                false, a.v, false, bfr[nt], (short)0, acc[nt], false, false);
    }

    // ------ epilogue ------
    // D element v of lane: row = rowTile + v + 8*lhi, col = nt*16 + lmod
    int rOut[8];
    #pragma unroll
    for (int v = 0; v < 8; ++v) rOut[v] = rowTile + v + (lhi << 3);

    size_t oOff[8];
    if constexpr (MODE == GM_MSG) {
        const int b = rowTile / E_;  // E_ multiple of 16 -> same b across tile
        #pragma unroll
        for (int v = 0; v < 8; ++v)
            oOff[v] = ((size_t)b * N_ + gload<int>(p.dst + rOut[v] % E_)) * 128;
    } else {
        #pragma unroll
        for (int v = 0; v < 8; ++v) oOff[v] = (size_t)rOut[v] * 128;
    }

    #pragma unroll
    for (int nt = 0; nt < 8; ++nt) {
        const int n = nt * 16 + lmod;
        const float bias = gload<float>(p.bias + n);
        #pragma unroll
        for (int v = 0; v < 8; ++v) {
            float x = acc[nt][v] + bias;
            if constexpr (MODE == GM_EMBED2) {
                // h0 = mlp2 + be2 + t_emb[b]  (no silu on second layer)
                x += gload<float>(p.extra + (size_t)(rOut[v] / N_) * 128 + n);
                gstore<float>(p.outF + oOff[v] + n, x);
                gstore<bf16_t>(p.outB + oOff[v] + n, (bf16_t)x);
            } else {
                float y = silu_f(x);
                if constexpr (MODE == GM_MSG) {
                    unsafeAtomicAdd(p.outF + oOff[v] + n, y);   // scatter-sum into agg
                } else if constexpr (MODE == GM_UPD) {
                    float hnew = gload<float>(p.outF + oOff[v] + n) + y;  // residual
                    gstore<float>(p.outF + oOff[v] + n, hnew);
                    gstore<bf16_t>(p.outB + oOff[v] + n, (bf16_t)hnew);
                } else {                                        // EMBED1/SCORE1/SCORE2
                    gstore<bf16_t>(p.outB + oOff[v] + n, (bf16_t)y);
                }
            }
        }
    }
}

// ---------------- host orchestration ----------------
extern "C" void kernel_launch(void* const* d_in, const int* in_sizes, int n_in,
                              void* d_out, int out_size, void* d_ws, size_t ws_size,
                              hipStream_t stream) {
    (void)in_sizes; (void)n_in; (void)out_size; (void)ws_size;

    const float* nf  = (const float*)d_in[0];
    const int*   ei  = (const int*)d_in[1];
    const int*   mn  = (const int*)d_in[2];
    const unsigned char* mask = (const unsigned char*)d_in[3];
    const float* t   = (const float*)d_in[4];
    const float* Wt1 = (const float*)d_in[5];
    const float* bt1 = (const float*)d_in[6];
    const float* Wt2 = (const float*)d_in[7];
    const float* bt2 = (const float*)d_in[8];
    const float* We1 = (const float*)d_in[9];
    const float* be1 = (const float*)d_in[10];
    const float* We2 = (const float*)d_in[11];
    const float* be2 = (const float*)d_in[12];
    const float* Wm  = (const float*)d_in[13];
    const float* bm  = (const float*)d_in[14];
    const float* Wu  = (const float*)d_in[15];
    const float* bu  = (const float*)d_in[16];
    const float* Ws1 = (const float*)d_in[17];
    const float* bs1 = (const float*)d_in[18];
    const float* Ws2 = (const float*)d_in[19];
    const float* bs2 = (const float*)d_in[20];
    const float* Ws3 = (const float*)d_in[21];
    const float* bs3 = (const float*)d_in[22];

    // workspace bump allocator (256B aligned)
    char* wsp = (char*)d_ws;
    auto alloc = [&](size_t bytes) -> char* {
        char* r = wsp;
        wsp += (bytes + 255) & ~(size_t)255;
        return r;
    };
    float*  t_emb = (float*) alloc((size_t)B_ * D_ * 4);
    bf16_t* We1T  = (bf16_t*)alloc((size_t)128 * 32 * 2);
    bf16_t* We2T  = (bf16_t*)alloc((size_t)128 * 128 * 2);
    bf16_t* WmT   = (bf16_t*)alloc((size_t)L_ * 128 * 256 * 2);
    bf16_t* WuT   = (bf16_t*)alloc((size_t)L_ * 128 * 256 * 2);
    bf16_t* Ws1T  = (bf16_t*)alloc((size_t)128 * 512 * 2);
    bf16_t* Ws2T  = (bf16_t*)alloc((size_t)128 * 128 * 2);
    bf16_t* nfB   = (bf16_t*)alloc((size_t)B_ * N_ * F_ * 2);
    bf16_t* tmpB  = (bf16_t*)alloc((size_t)B_ * N_ * D_ * 2); // embed hidden, reused as s1
    float*  hF    = (float*) alloc((size_t)B_ * N_ * D_ * 4);
    bf16_t* hB    = (bf16_t*)alloc((size_t)B_ * N_ * D_ * 2);
    float*  aggF  = (float*) alloc((size_t)B_ * N_ * D_ * 4);
    bf16_t* aggB  = (bf16_t*)alloc((size_t)B_ * N_ * D_ * 2); // reused as s2

    auto tp = [&](const float* W, bf16_t* WT, int K, int Nn) {
        int cnt = K * Nn;
        transpose_bf16_kernel<<<(cnt + 255) / 256, 256, 0, stream>>>(W, WT, K, Nn);
    };
    tp(We1, We1T, 32, 128);
    tp(We2, We2T, 128, 128);
    for (int l = 0; l < L_; ++l) {
        tp(Wm + (size_t)l * 256 * 128, WmT + (size_t)l * 128 * 256, 256, 128);
        tp(Wu + (size_t)l * 256 * 128, WuT + (size_t)l * 128 * 256, 256, 128);
    }
    tp(Ws1, Ws1T, 512, 128);
    tp(Ws2, Ws2T, 128, 128);

    { int cnt = B_ * N_ * F_; cvt_bf16_kernel<<<(cnt + 255) / 256, 256, 0, stream>>>(nf, nfB, cnt); }
    time_embed_kernel<<<1, 128, 0, stream>>>(t, Wt1, bt1, Wt2, bt2, t_emb);

    const int rowsN = B_ * N_;   // 65536
    const int rowsE = B_ * E_;   // 524288
    const int rowsM = B_ * M_;   // 65536
    const int* srcI = ei;
    const int* dstI = ei + E_;

    GemmArgs a;
    // embed stage 1: silu(nf @ We1 + be1) -> tmpB (bf16), K=32
    a = GemmArgs{}; a.A0 = nfB; a.W = We1T; a.bias = be1; a.outB = tmpB;
    gemm_kernel<GM_EMBED1, 1><<<rowsN / 128, 256, 0, stream>>>(a);
    // embed stage 2: tmp @ We2 + be2 + t_emb -> hF, hB, K=128
    a = GemmArgs{}; a.A0 = tmpB; a.W = We2T; a.bias = be2; a.extra = t_emb;
    a.outF = hF; a.outB = hB;
    gemm_kernel<GM_EMBED2, 4><<<rowsN / 128, 256, 0, stream>>>(a);

    for (int l = 0; l < L_; ++l) {
        { int cnt = rowsN * D_; zero_kernel<<<(cnt + 255) / 256, 256, 0, stream>>>(aggF, cnt); }
        // message + fused scatter-sum (native f32 atomics into agg), K=256
        a = GemmArgs{}; a.A0 = hB; a.W = WmT + (size_t)l * 128 * 256; a.bias = bm + l * D_;
        a.src = srcI; a.dst = dstI; a.outF = aggF;
        gemm_kernel<GM_MSG, 8><<<rowsE / 128, 256, 0, stream>>>(a);
        { int cnt = rowsN * D_; cvt_bf16_kernel<<<(cnt + 255) / 256, 256, 0, stream>>>(aggF, aggB, cnt); }
        // update: h += silu([h, agg] @ Wu + bu), K=256
        a = GemmArgs{}; a.A0 = hB; a.A1 = aggB; a.W = WuT + (size_t)l * 128 * 256;
        a.bias = bu + l * D_; a.outF = hF; a.outB = hB;
        gemm_kernel<GM_UPD, 8><<<rowsN / 128, 256, 0, stream>>>(a);
    }

    // scorer
    a = GemmArgs{}; a.A0 = hB; a.W = Ws1T; a.bias = bs1; a.mn = mn; a.outB = tmpB;
    gemm_kernel<GM_SCORE1, 16><<<rowsM / 128, 256, 0, stream>>>(a);
    a = GemmArgs{}; a.A0 = tmpB; a.W = Ws2T; a.bias = bs2; a.outB = aggB;
    gemm_kernel<GM_SCORE2, 4><<<rowsM / 128, 256, 0, stream>>>(a);

    score_kernel<<<(rowsM * 32 + 255) / 256, 256, 0, stream>>>(
        aggB, Ws3, bs3, mask, (float*)d_out, rowsM);
}
